// Attention_16449724745449
// MI455X (gfx1250) — matrix operand order repaired and checked by reading, not compiled, not run on hardware
//
#include <hip/hip_runtime.h>
#include <stdint.h>

// ---------------------------------------------------------------------------
// MI455X (gfx1250) fused attention layer.
// All matmuls run on v_wmma_f32_16x16x32_bf16 (bf16 in, fp32 accumulate):
// CDNA5's widest-K floating WMMA above fp8. fp32 WMMA is only 16x16x4, so
// bf16 is the throughput point for an fp32 reference on this chip.
// GEMMs are templated on (N, K) so all stride math folds into the 24-bit
// immediate offsets of global/ds ops instead of v_mul_u64 chains.
// ---------------------------------------------------------------------------

typedef unsigned short u16;
typedef __attribute__((ext_vector_type(16))) __bf16 v16bf;
typedef __attribute__((ext_vector_type(8)))  float  v8f;

union Frag16 {            // one WMMA 16-bit A/B operand: 16 bf16 = 8 VGPRs
  v16bf v;
  uint4 q[2];
};

__device__ __forceinline__ u16 f2bf(float f) {           // RNE f32 -> bf16
  unsigned int u = __float_as_uint(f);
  u += 0x7FFFu + ((u >> 16) & 1u);
  return (u16)(u >> 16);
}

__device__ __forceinline__ v8f wmma_bf16(const v16bf& a, const v16bf& b, const v8f& c) {
  return __builtin_amdgcn_wmma_f32_16x16x32_bf16(false, a, false, b, (short)0, c, false, false);
}

__device__ __forceinline__ float row_max16(float v) {
  #pragma unroll
  for (int m = 1; m < 16; m <<= 1) v = fmaxf(v, __shfl_xor(v, m, 32));
  return v;   // masks < 16 keep lanes 0-15 / 16-31 independent (wave32)
}
__device__ __forceinline__ float row_sum16(float v) {
  #pragma unroll
  for (int m = 1; m < 16; m <<= 1) v += __shfl_xor(v, m, 32);
  return v;
}

constexpr int   cB    = 2;
constexpr int   cL    = 2048;
constexpr int   cD    = 2048;
constexpr int   cH    = 8;
constexpr int   cKVH  = 4;
constexpr int   cHD   = 256;
constexpr int   cBL   = cB * cL;      // 4096 token rows
constexpr int   cQD   = cH * cHD;     // 2048
constexpr int   cKD   = cKVH * cHD;   // 1024
constexpr int   cWIN  = 1024;
constexpr float cKMASK = -2.3819763e38f;

// ---------------------------------------------------------------------------
// fp32 -> bf16 bulk convert
// ---------------------------------------------------------------------------
__global__ __launch_bounds__(256) void to_bf16_k(const float* __restrict__ in,
                                                 u16* __restrict__ out, size_t n) {
  size_t i = (size_t)blockIdx.x * blockDim.x + threadIdx.x;
  size_t stride = (size_t)gridDim.x * blockDim.x;
  for (; i < n; i += stride) out[i] = f2bf(in[i]);
}

// ---------------------------------------------------------------------------
// Tiled GEMM: C[M,N](f32) = A[M,K](bf16,row) * B[K,N](bf16,row)
// Block 256 thr = 8 waves. Block tile 128x128, wave tile 64x32, K step 32.
// B tile stored transposed in LDS so both fragments are ds_load_b128 pairs
// matching the 16-bit A(16x32)/B(32x16) WMMA VGPR layouts.
// N, K compile-time: address math folds into immediate offsets.
// ---------------------------------------------------------------------------
template <int N, int K>
__global__ __launch_bounds__(256) void gemm_bf16(const u16* __restrict__ A,
                                                 const u16* __restrict__ Bm,
                                                 float* __restrict__ C) {
  __shared__ u16 la[128][40];    // 128 rows x 32 k (+8 pad)
  __shared__ u16 lbT[128][40];   // 128 cols x 32 k (+8 pad), transposed B

  const int tid  = threadIdx.x;
  const int wave = tid >> 5, lane = tid & 31;
  const int l16  = lane & 15,  lh  = lane >> 4;
  const int wm   = wave >> 2;            // 0..1
  const int wn   = wave & 3;             // 0..3
  const int bm   = blockIdx.y * 128;
  const int bn   = blockIdx.x * 128;
  const int mbase = wm * 64;
  const int nbase = wn * 32;

  v8f acc[4][2];
  #pragma unroll
  for (int mt = 0; mt < 4; ++mt)
    #pragma unroll
    for (int nt = 0; nt < 2; ++nt)
      acc[mt][nt] = v8f{0.f,0.f,0.f,0.f,0.f,0.f,0.f,0.f};

  // per-thread tile-load coordinates (constant across k-loop)
  const int ar = tid >> 2, as = tid & 3;     // A: row 0..63(+64), 8-col seg
  const int br = tid >> 4, bs = tid & 15;    // B: k row 0..15(+16), 8-col seg
  const u16* agp0 = A + (size_t)(bm + ar) * K + as * 8;
  const u16* agp1 = A + (size_t)(bm + ar + 64) * K + as * 8;
  const u16* bgp0 = Bm + (size_t)br * N + bn + bs * 8;
  const u16* bgp1 = Bm + (size_t)(br + 16) * N + bn + bs * 8;

  for (int k0 = 0; k0 < K; k0 += 32) {
    // A tile 128x32
    *(uint4*)&la[ar][as * 8]      = *(const uint4*)(agp0 + k0);
    *(uint4*)&la[ar + 64][as * 8] = *(const uint4*)(agp1 + k0);
    // B tile 32x128, transposed into LDS
    {
      uint4 d0 = *(const uint4*)(bgp0 + (size_t)k0 * N);
      uint4 d1 = *(const uint4*)(bgp1 + (size_t)k0 * N);
      const u16* t0 = (const u16*)&d0;
      const u16* t1 = (const u16*)&d1;
      #pragma unroll
      for (int e = 0; e < 8; ++e) {
        lbT[bs * 8 + e][br]      = t0[e];
        lbT[bs * 8 + e][br + 16] = t1[e];
      }
    }
    __syncthreads();

    Frag16 af[4], bf[2];
    #pragma unroll
    for (int mt = 0; mt < 4; ++mt) {
      int row = mbase + mt * 16 + l16;   // A: k = (e/8)*16 + lh*8 + e%8
      af[mt].q[0] = *(const uint4*)&la[row][lh * 8];
      af[mt].q[1] = *(const uint4*)&la[row][16 + lh * 8];
    }
    #pragma unroll
    for (int nt = 0; nt < 2; ++nt) {
      int col = nbase + nt * 16 + l16;   // B: k = (e/8)*16 + lh*8 + e%8
      bf[nt].q[0] = *(const uint4*)&lbT[col][lh * 8];
      bf[nt].q[1] = *(const uint4*)&lbT[col][16 + lh * 8];
    }
    #pragma unroll
    for (int mt = 0; mt < 4; ++mt)
      #pragma unroll
      for (int nt = 0; nt < 2; ++nt)
        acc[mt][nt] = wmma_bf16(af[mt].v, bf[nt].v, acc[mt][nt]);
    __syncthreads();
  }

  // C layout: m = r + 8*(lane/16), n = lane%16. Single base + imm offsets.
  float* cp = C + (size_t)(bm + mbase + 8 * lh) * N + bn + nbase + l16;
  #pragma unroll
  for (int mt = 0; mt < 4; ++mt)
    #pragma unroll
    for (int nt = 0; nt < 2; ++nt)
      #pragma unroll
      for (int r = 0; r < 8; ++r)
        cp[(size_t)(mt * 16 + r) * N + nt * 16] = acc[mt][nt][r];
}

// ---------------------------------------------------------------------------
// Per-(token, head) RMS-norm + RoPE, emits bf16. 256 threads = head_dim.
// ---------------------------------------------------------------------------
__global__ __launch_bounds__(256) void rmsnorm_rope(const float* __restrict__ X,
                                                    const float* __restrict__ scale,
                                                    const int* __restrict__ positions,
                                                    u16* __restrict__ Out, int nheads) {
  const int t = blockIdx.x;            // token 0..BL-1
  const int h = blockIdx.y;
  const int d = threadIdx.x;           // 0..255
  const int wv = d >> 5, ln = d & 31;
  const float* xp = X + ((size_t)t * nheads + h) * cHD;

  float x = xp[d];
  float ss = x * x;
  #pragma unroll
  for (int m = 1; m < 32; m <<= 1) ss += __shfl_xor(ss, m, 32);
  __shared__ float red[8];
  if (ln == 0) red[wv] = ss;
  __syncthreads();
  float tot = 0.f;
  #pragma unroll
  for (int i = 0; i < 8; ++i) tot += red[i];
  const float rr = rsqrtf(tot * (1.0f / cHD) + 1e-6f);
  const float nv = x * rr * scale[d];

  __shared__ float sh[256];
  sh[d] = nv;
  __syncthreads();

  const int   pos = positions[t];
  const int   dr  = d & 127;
  const float fr  = powf(10000.0f, (2.0f * dr) * (1.0f / cHD));
  const float tt  = (float)pos / fr;
  const float c   = cosf(tt), sn = sinf(tt);
  const float x1  = sh[dr], x2 = sh[dr + 128];
  const float ov  = (d < 128) ? (x1 * c - x2 * sn) : (x2 * c + x1 * sn);
  Out[((size_t)t * nheads + h) * cHD + d] = f2bf(ov);
}

// ---------------------------------------------------------------------------
// V fp32 [B,L,KVH,HD] -> bf16 transposed [B,KVH,HD,L] (PV B-fragments become
// two contiguous 16B loads per lane).
// ---------------------------------------------------------------------------
__global__ __launch_bounds__(256) void convert_v(const float* __restrict__ V,
                                                 u16* __restrict__ Vt) {
  size_t i = (size_t)blockIdx.x * blockDim.x + threadIdx.x;
  const size_t total = (size_t)cBL * cKD;
  if (i >= total) return;
  int d   = (int)(i & 255);
  size_t r = i >> 8;
  int kvh = (int)(r & 3);  r >>= 2;
  int l   = (int)(r & (cL - 1));
  int b   = (int)(r >> 11);
  Vt[(((size_t)b * cKVH + kvh) * cHD + d) * cL + l] = f2bf(V[i]);
}

// ---------------------------------------------------------------------------
// Windowed flash attention. Block = 128 thr = 4 waves; each wave owns a 16-row
// Q tile (grid.x covers L/64). Online softmax; soft-cap tanh; causal+window
// mask; P goes C-layout -> A-layout through per-wave LDS (s_wait_dscnt 0).
// global_prefetch_b8 pulls the next key block while WMMA runs.
// ---------------------------------------------------------------------------
__global__ __launch_bounds__(128) void attn_fwd(const u16* __restrict__ Qb,
                                                const u16* __restrict__ Kb,
                                                const u16* __restrict__ Vt,
                                                u16* __restrict__ Ob) {
  __shared__ u16 pbuf[4][16][32];   // per-wave P transpose buffer

  const int tid  = threadIdx.x;
  const int wave = tid >> 5, lane = tid & 31;
  const int l16  = lane & 15,  lh  = lane >> 4;
  const int q0   = blockIdx.x * 64 + wave * 16;
  const int h    = blockIdx.y;
  const int b    = blockIdx.z;
  const int kvh  = h >> 1;          // GROUPS = 2

  // Q fragments: 8 chunks of 32 dims (A layout: row = l16, k by lh/e)
  Frag16 qf[8];
  {
    const u16* qrow = Qb + ((size_t)(b * cL + q0 + l16) * cH + h) * cHD;
    #pragma unroll
    for (int c = 0; c < 8; ++c) {
      qf[c].q[0] = *(const uint4*)(qrow + c * 32 + lh * 8);
      qf[c].q[1] = *(const uint4*)(qrow + c * 32 + 16 + lh * 8);
    }
  }

  v8f o[16];
  #pragma unroll
  for (int d = 0; d < 16; ++d) o[d] = v8f{0.f,0.f,0.f,0.f,0.f,0.f,0.f,0.f};
  float mrow[8], lrow[8];
  #pragma unroll
  for (int r = 0; r < 8; ++r) { mrow[r] = cKMASK; lrow[r] = 0.f; }

  int s_begin = q0 - (cWIN - 1);
  if (s_begin < 0) s_begin = 0;
  s_begin &= ~31;
  const int s_last = q0 + 15;

  for (int s = s_begin; s <= s_last; s += 32) {
    // ---- S = Q @ K^T over 32 keys (two 16-col tiles) ----
    v8f s0 = v8f{0.f,0.f,0.f,0.f,0.f,0.f,0.f,0.f};
    v8f s1 = s0;
    const int k0i = min(s + l16,      cL - 1);
    const int k1i = min(s + 16 + l16, cL - 1);
    const u16* kr0 = Kb + ((size_t)(b * cL + k0i) * cKVH + kvh) * cHD;
    const u16* kr1 = Kb + ((size_t)(b * cL + k1i) * cKVH + kvh) * cHD;

    // prefetch next key block (K rows are +32 tokens = +32*KVH*HD elems)
    if (s + 32 <= s_last) {
      __builtin_prefetch(kr0 + 32 * cKVH * cHD, 0, 0);
      __builtin_prefetch(kr1 + 32 * cKVH * cHD, 0, 0);
    }

    #pragma unroll
    for (int c = 0; c < 8; ++c) {
      Frag16 kf0, kf1;                 // B layout: n = key(l16), k = (e/8)*16 + lh*8 + e%8
      kf0.q[0] = *(const uint4*)(kr0 + c * 32 + lh * 8);
      kf0.q[1] = *(const uint4*)(kr0 + c * 32 + 16 + lh * 8);
      kf1.q[0] = *(const uint4*)(kr1 + c * 32 + lh * 8);
      kf1.q[1] = *(const uint4*)(kr1 + c * 32 + 16 + lh * 8);
      s0 = wmma_bf16(qf[c].v, kf0.v, s0);
      s1 = wmma_bf16(qf[c].v, kf1.v, s1);
    }

    // ---- soft-cap + causal/window mask (C layout: row=r+8*lh, col=l16) ----
    #pragma unroll
    for (int r = 0; r < 8; ++r) {
      const int qr = q0 + r + 8 * lh;
      const int key0 = s + l16, key1 = s + 16 + l16;
      const bool v0 = (key0 <= qr) && (key0 > qr - cWIN);
      const bool v1 = (key1 <= qr) && (key1 > qr - cWIN);
      s0[r] = v0 ? 50.0f * tanhf(s0[r] * 0.02f) : cKMASK;
      s1[r] = v1 ? 50.0f * tanhf(s1[r] * 0.02f) : cKMASK;
    }

    // ---- online softmax update ----
    #pragma unroll
    for (int r = 0; r < 8; ++r) {
      const float tm = row_max16(fmaxf(s0[r], s1[r]));
      const float mn = fmaxf(mrow[r], tm);
      const float sc = __expf(mrow[r] - mn);
      mrow[r] = mn;
      const float p0 = (s0[r] > -1e30f) ? __expf(s0[r] - mn) : 0.f;
      const float p1 = (s1[r] > -1e30f) ? __expf(s1[r] - mn) : 0.f;
      lrow[r] = lrow[r] * sc + row_sum16(p0 + p1);
      pbuf[wave][r + 8 * lh][l16]      = f2bf(p0);
      pbuf[wave][r + 8 * lh][16 + l16] = f2bf(p1);
      #pragma unroll
      for (int d = 0; d < 16; ++d) o[d][r] *= sc;
    }

    // wave-private LDS RAW: wait DS ops before re-reading as A-fragment
    asm volatile("s_wait_dscnt 0" ::: "memory");

    Frag16 pf;                         // A layout: row=l16, k = lh*16 + e
    pf.q[0] = *(const uint4*)&pbuf[wave][l16][lh * 16];
    pf.q[1] = *(const uint4*)&pbuf[wave][l16][lh * 16 + 8];

    // ---- O += P @ V : 16 dim tiles ----
    const int sv = min(s + lh * 16, cL - 16);
    const u16* vb = Vt + ((size_t)(b * cKVH + kvh) * cHD) * cL + sv;
    if (s + 32 <= s_last) __builtin_prefetch(vb + 32, 0, 0);
    #pragma unroll
    for (int d = 0; d < 16; ++d) {
      Frag16 vf;                       // B layout: n = dim(l16), k = key(lh*16+e)
      const u16* vp = vb + (size_t)(d * 16 + l16) * cL;
      vf.q[0] = *(const uint4*)(vp);
      vf.q[1] = *(const uint4*)(vp + 8);
      o[d] = wmma_bf16(pf.v, vf.v, o[d]);
    }
  }

  // ---- normalize and store bf16 [B,L,H,HD] for the output projection ----
  float inv[8];
  #pragma unroll
  for (int r = 0; r < 8; ++r) inv[r] = 1.0f / lrow[r];
  u16* ob = Ob + ((size_t)(b * cL + q0 + 8 * lh) * cH + h) * cHD + l16;
  #pragma unroll
  for (int d = 0; d < 16; ++d)
    #pragma unroll
    for (int r = 0; r < 8; ++r)
      ob[(size_t)r * cH * cHD + d * 16] = f2bf(o[d][r] * inv[r]);
}

// ---------------------------------------------------------------------------
// Host-side orchestration
// ---------------------------------------------------------------------------
extern "C" void kernel_launch(void* const* d_in, const int* in_sizes, int n_in,
                              void* d_out, int out_size, void* d_ws, size_t ws_size,
                              hipStream_t stream) {
  (void)in_sizes; (void)n_in; (void)out_size; (void)ws_size;

  const float* x         = (const float*)d_in[0];
  const int*   positions = (const int*)d_in[1];
  // d_in[2] attn_mask: causal tril — recomputed analytically in-kernel
  const float* wq      = (const float*)d_in[3];
  const float* wk      = (const float*)d_in[4];
  const float* wv      = (const float*)d_in[5];
  const float* wo      = (const float*)d_in[6];
  const float* q_scale = (const float*)d_in[7];
  const float* k_scale = (const float*)d_in[8];
  float* out = (float*)d_out;

  char* ws = (char*)d_ws;
  auto alloc = [&](size_t bytes) -> char* {
    char* p = ws;
    ws += (bytes + 255) & ~(size_t)255;
    return p;
  };

  const size_t nx  = (size_t)cBL * cD;
  const size_t nwq = (size_t)cD * cQD;
  const size_t nwk = (size_t)cD * cKD;
  const size_t nwo = (size_t)cQD * cD;

  u16*   xb  = (u16*)alloc(nx * 2);
  u16*   wqb = (u16*)alloc(nwq * 2);
  u16*   wkb = (u16*)alloc(nwk * 2);
  u16*   wvb = (u16*)alloc(nwk * 2);
  u16*   wob = (u16*)alloc(nwo * 2);
  float* Qf  = (float*)alloc((size_t)cBL * cQD * 4);
  float* Kf  = (float*)alloc((size_t)cBL * cKD * 4);
  float* Vf  = (float*)alloc((size_t)cBL * cKD * 4);
  u16*   Qn  = (u16*)alloc((size_t)cBL * cQD * 2);
  u16*   Kn  = (u16*)alloc((size_t)cBL * cKD * 2);
  u16*   Vt  = (u16*)alloc((size_t)cBL * cKD * 2);
  u16*   Ab  = (u16*)alloc((size_t)cBL * cQD * 2);

  // 1) bf16 conversions of activations + weights
  to_bf16_k<<<2048, 256, 0, stream>>>(x,  xb,  nx);
  to_bf16_k<<<2048, 256, 0, stream>>>(wq, wqb, nwq);
  to_bf16_k<<<1024, 256, 0, stream>>>(wk, wkb, nwk);
  to_bf16_k<<<1024, 256, 0, stream>>>(wv, wvb, nwk);
  to_bf16_k<<<2048, 256, 0, stream>>>(wo, wob, nwo);

  // 2) QKV projections (WMMA GEMMs)
  gemm_bf16<cQD, cD><<<dim3(cQD / 128, cBL / 128), 256, 0, stream>>>(xb, wqb, Qf);
  gemm_bf16<cKD, cD><<<dim3(cKD / 128, cBL / 128), 256, 0, stream>>>(xb, wkb, Kf);
  gemm_bf16<cKD, cD><<<dim3(cKD / 128, cBL / 128), 256, 0, stream>>>(xb, wvb, Vf);

  // 3) RMS-norm + RoPE -> bf16 Q/K; V -> bf16 transposed
  rmsnorm_rope<<<dim3(cBL, cH),   256, 0, stream>>>(Qf, q_scale, positions, Qn, cH);
  rmsnorm_rope<<<dim3(cBL, cKVH), 256, 0, stream>>>(Kf, k_scale, positions, Kn, cKVH);
  convert_v<<<(cBL * cKD + 255) / 256, 256, 0, stream>>>(Vf, Vt);

  // 4) windowed flash attention (WMMA QK^T and PV)
  attn_fwd<<<dim3(cL / 64, cH, cB), 128, 0, stream>>>(Qn, Kn, Vt, Ab);

  // 5) output projection (WMMA GEMM) -> fp32 result
  gemm_bf16<cD, cQD><<<dim3(cD / 128, cBL / 128), 256, 0, stream>>>(Ab, wob, out);
}
